// modulated_conv2d_45500883534246
// MI455X (gfx1250) — matrix operand order repaired and checked by reading, not compiled, not run on hardware
//
#include <hip/hip_runtime.h>
#include <math.h>

typedef __attribute__((ext_vector_type(16))) __bf16 v16bf;
typedef __attribute__((ext_vector_type(8)))  __bf16 v8bf;
typedef __attribute__((ext_vector_type(2)))  __bf16 v2bf;
typedef __attribute__((ext_vector_type(8)))  float  v8f;

#define CIN   512
#define COUT  512
#define HWDIM 64
#define BATCH 16

// equalized-LR coefficients (gain=1, lr_mul=1)
#define CONV_COEF 0.014731391274719738f   // 1/sqrt(3*3*512)
#define FC_COEF   0.044194173824159216f   // 1/sqrt(512)

// ---------------------------------------------------------------------------
// 1) style: s[b,i] = (w_latents[b,:] @ fc_w[:,i]) * fc_coef + fc_b[i] + 1
// ---------------------------------------------------------------------------
__global__ void style_k(const float* __restrict__ wl, const float* __restrict__ fcw,
                        const float* __restrict__ fcb, float* __restrict__ s) {
  const int b = blockIdx.x;
  const int i = threadIdx.x;
  const float* wlr = wl + b * 512;
  float acc = 0.f;
#pragma unroll 4
  for (int j = 0; j < 512; ++j) acc += wlr[j] * fcw[j * 512 + i];
  s[b * 512 + i] = acc * FC_COEF + fcb[i] + 1.0f;
}

// ---------------------------------------------------------------------------
// 2) wsq[i,o] = sum_{kh,kw} w[kh,kw,i,o]^2
// ---------------------------------------------------------------------------
__global__ void wsq_k(const float* __restrict__ w, float* __restrict__ wsq) {
  const int idx = blockIdx.x * 256 + threadIdx.x;   // 512*512
  const int o = idx & 511;
  const int i = idx >> 9;
  float acc = 0.f;
#pragma unroll
  for (int kk = 0; kk < 9; ++kk) {
    float v = w[(kk * 512 + i) * 512 + o];
    acc += v * v;
  }
  wsq[i * 512 + o] = acc;
}

// ---------------------------------------------------------------------------
// 3) wt[kk][o][i] = bf16(w[kh,kw,i,o] * conv_coef)   (transposed, bf16)
// ---------------------------------------------------------------------------
__global__ void wt_k(const float* __restrict__ w, __bf16* __restrict__ wt) {
  __shared__ float tile[32][33];
  const int kk = blockIdx.z;
  const int i0 = blockIdx.x * 32, o0 = blockIdx.y * 32;
  const int tx = threadIdx.x, ty = threadIdx.y;
  tile[ty][tx] = w[(kk * 512 + i0 + ty) * 512 + (o0 + tx)];
  __syncthreads();
  wt[(size_t)(kk * 512 + o0 + ty) * 512 + (i0 + tx)] = (__bf16)(tile[tx][ty] * CONV_COEF);
}

// ---------------------------------------------------------------------------
// 4) d[b,o] = rsqrt(cc^2 * sum_i s[b,i]^2 * wsq[i,o] + 1e-8)
// ---------------------------------------------------------------------------
__global__ void demod_k(const float* __restrict__ s, const float* __restrict__ wsq,
                        float* __restrict__ d) {
  const int b = blockIdx.x;
  const int o = threadIdx.x;
  const float* sr = s + b * 512;
  float acc = 0.f;
#pragma unroll 4
  for (int i = 0; i < 512; ++i) {
    float sv = sr[i];
    acc += sv * sv * wsq[i * 512 + o];
  }
  d[b * 512 + o] = rsqrtf(CONV_COEF * CONV_COEF * acc + 1e-8f);
}

// ---------------------------------------------------------------------------
// 5) xb[b,h,w,i] = bf16(x[b,h,w,i] * s[b,i])
// ---------------------------------------------------------------------------
__global__ void xmod_k(const float* __restrict__ x, const float* __restrict__ s,
                       __bf16* __restrict__ xb) {
  const long long gid = (long long)blockIdx.x * blockDim.x + threadIdx.x;
  const long long e = gid * 2;
  const int i = (int)(e & 511);
  const int b = (int)(e >> 21);                      // 64*64*512 = 2^21 per image
  const float2 xv = *(const float2*)(x + e);
  const float s0 = s[b * 512 + i];
  const float s1 = s[b * 512 + i + 1];
  v2bf rv;
  rv.x = (__bf16)(xv.x * s0);
  rv.y = (__bf16)(xv.y * s1);
  *(v2bf*)(xb + e) = rv;
}

// ---------------------------------------------------------------------------
// async global -> LDS (CDNA5, ASYNCcnt). Per ISA 08 4.4, IOFFSET is added to
// BOTH the LDS destination and the global source address, so consecutive
// 16B chunks need only one base VGPR (pair) + an immediate offset.
// ---------------------------------------------------------------------------
#define ASYNC_B128(lds_base, g_base, OFF)                                     \
  asm volatile("global_load_async_to_lds_b128 %0, %1, off offset:" #OFF      \
               :: "v"(lds_base), "v"(g_base) : "memory")

__device__ __forceinline__ void wait_async0() {
  asm volatile("s_wait_asynccnt 0" ::: "memory");
}

// ---------------------------------------------------------------------------
// 6) main conv, implicit GEMM, v_wmma_f32_16x16x32_bf16
//    block = 256 thr (8 waves); tile = 64 pixels (one row) x 256 out-ch
//    K staged in chunks of 64 through double-buffered LDS:
//      B (weights) via global_load_async_to_lds_b128 (offset-folded chunks)
//      A (activations) via global->reg (zero-padded) -> ds_store
//    one s_wait_asynccnt + one barrier per stage; 16 WMMA / wave / stage
// ---------------------------------------------------------------------------
__launch_bounds__(256)
__global__ void conv_k(const __bf16* __restrict__ xb, const __bf16* __restrict__ wt,
                       const float* __restrict__ d, const float* __restrict__ bias,
                       float* __restrict__ out) {
  // row stride 72 elems = 144 B : 16B-aligned, conflict-free across 16 rows
  __shared__ __align__(16) __bf16 Abuf[2][64 * 72];    // 2 x  9.0 KB
  __shared__ __align__(16) __bf16 Bbuf[2][256 * 72];   // 2 x 36.0 KB

  const int tid = threadIdx.x;
  const int h   = blockIdx.x;          // output row 0..63
  const int o0  = blockIdx.y * 256;    // out-channel block
  const int b   = blockIdx.z;          // image

  const int lane = tid & 31;
  const int wid  = tid >> 5;
  const int mw   = wid & 1;            // M group (2 subtiles each)
  const int nw   = wid >> 1;           // N group (4 subtiles each)
  const int half = lane >> 4;          // K-half select per ISA layout
  const int l16  = lane & 15;

  const int apix  = tid >> 2;          // 0..63 (pixel in row)
  const int apart = tid & 3;           // 0..3  (8 bf16 each; x2 halves = 64 ch)
  const long long xbase = (long long)b * (HWDIM * HWDIM * CIN);

  v8f acc[8] = {};

  const int NSTAGE = 72;               // 9 taps * (512/64) channel chunks

  // ---- issue async B-tile loads (256 rows x 64 ch) into buffer buf
  auto issue_b = [&](int kk, int ics, int buf) {
    const void* g = (const void*)(wt + (size_t)(kk * 512 + o0 + tid) * CIN + ics);
    const unsigned l = (unsigned)(size_t)&Bbuf[buf][tid * 72];
    ASYNC_B128(l, g, 0);
    ASYNC_B128(l, g, 16);
    ASYNC_B128(l, g, 32);
    ASYNC_B128(l, g, 48);
    ASYNC_B128(l, g, 64);
    ASYNC_B128(l, g, 80);
    ASYNC_B128(l, g, 96);
    ASYNC_B128(l, g, 112);
  };

  // ---- A-tile: global -> regs with border zero-padding
  auto load_a = [&](int kh, int kwp, int ics, v8bf& a0, v8bf& a1) {
    const int hin = h + kh - 1;
    const int win = apix + kwp - 1;
    a0 = v8bf{};
    a1 = v8bf{};
    if (hin >= 0 && hin < HWDIM && win >= 0 && win < HWDIM) {
      const __bf16* p = xb + xbase + ((long long)hin * HWDIM + win) * CIN + ics + apart * 8;
      a0 = *(const v8bf*)p;
      a1 = *(const v8bf*)(p + 32);
    }
  };
  auto store_a = [&](int buf, v8bf a0, v8bf a1) {
    __bf16* p = &Abuf[buf][apix * 72 + apart * 8];
    *(v8bf*)p = a0;
    *(v8bf*)(p + 32) = a1;
  };

  // ---- 16 WMMAs on a staged 64-wide K chunk
  auto compute = [&](int buf) {
#pragma unroll
    for (int ks = 0; ks < 2; ++ks) {
      v16bf af[2];
#pragma unroll
      for (int si = 0; si < 2; ++si) {
        const __bf16* ap = &Abuf[buf][((mw * 2 + si) * 16 + l16) * 72 + ks * 32 + half * 8];
        v8bf lo = *(const v8bf*)ap;
        v8bf hi = *(const v8bf*)(ap + 16);
        af[si] = __builtin_shufflevector(lo, hi, 0,1,2,3,4,5,6,7,8,9,10,11,12,13,14,15);
      }
#pragma unroll
      for (int tj = 0; tj < 4; ++tj) {
        const __bf16* bp = &Bbuf[buf][((nw * 4 + tj) * 16 + l16) * 72 + ks * 32 + half * 8];
        v8bf lo = *(const v8bf*)bp;
        v8bf hi = *(const v8bf*)(bp + 16);
        v16bf bfr = __builtin_shufflevector(lo, hi, 0,1,2,3,4,5,6,7,8,9,10,11,12,13,14,15);
#pragma unroll
        for (int si = 0; si < 2; ++si)
          acc[si * 4 + tj] = __builtin_amdgcn_wmma_f32_16x16x32_bf16(
              false, af[si], false, bfr, (short)0, acc[si * 4 + tj], false, false);
      }
    }
  };

  // ---- software pipeline: prologue (stage 0: kh=0,kw=0,ics=0)
  v8bf a0, a1;
  load_a(0, 0, 0, a0, a1);
  issue_b(0, 0, 0);
  store_a(0, a0, a1);
  wait_async0();
  __syncthreads();

  // incrementally-maintained params of the NEXT stage (uniform scalars)
  int n_ics = 64, n_kw = 0, n_kh = 0;

  for (int st = 0; st < NSTAGE; ++st) {
    const int cur  = st & 1;
    const int nxt  = cur ^ 1;
    const bool more = (st + 1 < NSTAGE);
    if (more) {
      load_a(n_kh, n_kw, n_ics, a0, a1);          // global loads in flight during WMMA
      issue_b(n_kh * 3 + n_kw, n_ics, nxt);       // async B overlaps WMMA
      n_ics += 64;
      if (n_ics == CIN) {
        n_ics = 0;
        if (++n_kw == 3) { n_kw = 0; ++n_kh; }
      }
    }
    compute(cur);
    if (more) store_a(nxt, a0, a1);
    wait_async0();
    __syncthreads();
  }

  // ---- epilogue: y = acc * d[b,o] + bias[o]  (non-temporal, write-once)
#pragma unroll
  for (int tj = 0; tj < 4; ++tj) {
    const int n = o0 + (nw * 4 + tj) * 16 + l16;
    const float dd = d[b * 512 + n];
    const float bb = bias[n];
#pragma unroll
    for (int si = 0; si < 2; ++si) {
      const int wc0 = (mw * 2 + si) * 16 + half * 8;   // C layout: vgpr r -> M = r + half*8
#pragma unroll
      for (int r = 0; r < 8; ++r) {
        const long long oi = ((long long)b * 4096 + h * 64 + (wc0 + r)) * 512 + n;
        __builtin_nontemporal_store(acc[si * 4 + tj][r] * dd + bb, &out[oi]);
      }
    }
  }
}

// ---------------------------------------------------------------------------
extern "C" void kernel_launch(void* const* d_in, const int* in_sizes, int n_in,
                              void* d_out, int out_size, void* d_ws, size_t ws_size,
                              hipStream_t stream) {
  const float* x    = (const float*)d_in[0];   // [16,64,64,512]
  const float* wl   = (const float*)d_in[1];   // [16,512]
  const float* w    = (const float*)d_in[2];   // [3,3,512,512]
  const float* bias = (const float*)d_in[3];   // [512]
  const float* fcw  = (const float*)d_in[4];   // [512,512]
  const float* fcb  = (const float*)d_in[5];   // [512]
  float* out = (float*)d_out;

  // workspace layout (bytes): ~69.6 MB total
  char* ws = (char*)d_ws;
  float*  s_buf  = (float*)(ws + 0);              //  32 KB: s[16,512]
  float*  d_buf  = (float*)(ws + 32768);          //  32 KB: d[16,512]
  float*  wsq    = (float*)(ws + 65536);          //   1 MB: wsq[512,512]
  __bf16* wt     = (__bf16*)(ws + 1114112);       // 4.5 MB: wt[9,512,512] bf16
  __bf16* xbuf   = (__bf16*)(ws + 5832704);       //  64 MB: xb[16,64,64,512] bf16

  style_k<<<dim3(16),        dim3(512),    0, stream>>>(wl, fcw, fcb, s_buf);
  wsq_k  <<<dim3(1024),      dim3(256),    0, stream>>>(w, wsq);
  wt_k   <<<dim3(16, 16, 9), dim3(32, 32), 0, stream>>>(w, wt);
  demod_k<<<dim3(16),        dim3(512),    0, stream>>>(s_buf, wsq, d_buf);
  xmod_k <<<dim3(65536),     dim3(256),    0, stream>>>(x, s_buf, xbuf);
  conv_k <<<dim3(64, 2, 16), dim3(256),    0, stream>>>(xbuf, wt, d_buf, bias, out);
}
